// CrossBlock_86981677679342
// MI455X (gfx1250) — matrix operand order repaired and checked by reading, not compiled, not run on hardware
//
#include <hip/hip_runtime.h>
#include <math.h>

// ---------------------------------------------------------------------------
// CDNA5 (gfx1250) WMMA helpers: v_wmma_f32_16x16x32_bf16, wave32.
// ---------------------------------------------------------------------------
typedef __bf16 bf16;
typedef __attribute__((ext_vector_type(16))) __bf16 v16bf;
typedef __attribute__((ext_vector_type(8)))  float  v8f;

union BFrag { v16bf v; unsigned int u[8]; };

__device__ __forceinline__ unsigned lane_id() { return threadIdx.x & 31u; }

// Hardware f32->bf16 convert (RNE on gfx1250) instead of integer bit-math.
__device__ __forceinline__ unsigned pack2(float a, float b) {
  union { bf16 h[2]; unsigned u; } c;
  c.h[0] = (bf16)a;
  c.h[1] = (bf16)b;
  return c.u;
}
__device__ __forceinline__ void store_bf(bf16* p, float x) { *p = (bf16)x; }

// A fragment 16x32 bf16 (ISA §7.12.2 table): lane i=l&15 owns row i,
// half h=l>>4 selects K sub-blocks {0-7,16-23} vs {8-15,24-31}, 2 elems/VGPR.
// Per lane this is two contiguous 16B chunks -> compiler emits 2x b128 loads.
__device__ __forceinline__ v16bf load_a_bf16(const bf16* base, int ld) {
  unsigned l = lane_id(), i = l & 15u, h = l >> 4;
  const bf16* row = base + (size_t)i * ld;
  BFrag f;
#pragma unroll
  for (int v = 0; v < 8; ++v) {
    int k = ((v >> 2) << 4) + (int)(h << 3) + ((v & 3) << 1);
    f.u[v] = *(const unsigned int*)(row + k);
  }
  return f.v;
}

// B fragment 32x16 bf16: lane n=l&15 owns column n; same K order as A:
// half h=l>>4 selects K sub-blocks {0-7,16-23} vs {8-15,24-31}.
// Storage must be column-major (K contiguous), i.e. Wt[col][k].
__device__ __forceinline__ v16bf load_b_bf16(const bf16* base, int ld) {
  unsigned l = lane_id(), n = l & 15u, h = l >> 4;
  const bf16* col = base + (size_t)n * ld;
  BFrag f;
#pragma unroll
  for (int v = 0; v < 8; ++v) {
    int k = ((v >> 2) << 4) + (int)(h << 3) + ((v & 3) << 1);
    f.u[v] = *(const unsigned int*)(col + k);
  }
  return f.v;
}

__device__ __forceinline__ v8f wmma_bf16(v16bf a, v16bf b, v8f c) {
  return __builtin_amdgcn_wmma_f32_16x16x32_bf16(false, a, false, b, (short)0, c,
                                                 false, false);
}
__device__ __forceinline__ v8f zero8() {
  v8f z = {0.f, 0.f, 0.f, 0.f, 0.f, 0.f, 0.f, 0.f};
  return z;
}
__device__ __forceinline__ float gelu_exact(float v) {
  return 0.5f * v * (1.0f + erff(v * 0.70710678118654752f));
}

#define EMBED 256
#define HEADS 4
#define HD 64
#define BATCH 4

// ---------------------------------------------------------------------------
// Kernel 0: elementwise f32 -> bf16 (x pre-conversion), 4 elems/thread.
// ---------------------------------------------------------------------------
__global__ void k_cvt(const float* __restrict__ src, bf16* __restrict__ dst,
                      int n4) {
  int idx = blockIdx.x * blockDim.x + threadIdx.x;
  if (idx >= n4) return;
  const float4 v = ((const float4*)src)[idx];
  unsigned* d = (unsigned*)dst + (size_t)idx * 2;
  d[0] = pack2(v.x, v.y);
  d[1] = pack2(v.z, v.w);
}

// ---------------------------------------------------------------------------
// Kernel 1: weight convert + transpose. W[K][NC] f32 -> Wt[NC][K] bf16.
// ---------------------------------------------------------------------------
__global__ void k_wt(const float* __restrict__ w, bf16* __restrict__ wt,
                     int K, int NC) {
  int idx = blockIdx.x * blockDim.x + threadIdx.x;
  if (idx >= K * NC) return;
  int k = idx % K, c = idx / K;
  store_bf(wt + idx, w[(size_t)k * NC + c]);
}

// ---------------------------------------------------------------------------
// Kernel 2: fused QK/V projection. xb[B*N][256] bf16 -> qk[B,H,N,HD] bf16 and
// v^T[B,H,HD,N] bf16 (transposed so the P·V B-fragment reads K-contiguous).
// block 128 = 4 waves, wave = 16 rows; grid.y splits 256 cols into 4x64.
// All 8 B-fragments of a K-chunk preloaded -> loads clause together and
// WMMAs retire against staggered s_wait_loadcnt instead of 0 each time.
// ---------------------------------------------------------------------------
__global__ void k_project(const bf16* __restrict__ xb,
                          const bf16* __restrict__ wqkt, const float* __restrict__ bqk,
                          const bf16* __restrict__ wvt,  const float* __restrict__ bv,
                          bf16* __restrict__ qk_out, bf16* __restrict__ vt_out,
                          int N) {
  int wave = threadIdx.x >> 5;
  int row0 = (blockIdx.x * 4 + wave) * 16;
  int col0 = blockIdx.y * 64;
  // 16-row tile never crosses a batch (N % 16 == 0): one div per wave.
  int bidx = row0 / N;
  int nbase = row0 - bidx * N;

  v8f accq[4], accv[4];
#pragma unroll
  for (int t = 0; t < 4; ++t) { accq[t] = zero8(); accv[t] = zero8(); }

  for (int kc = 0; kc < EMBED; kc += 32) {
    v16bf a = load_a_bf16(xb + (size_t)row0 * EMBED + kc, EMBED);
    v16bf bq[4], bv4[4];
#pragma unroll
    for (int t = 0; t < 4; ++t) {
      bq[t]  = load_b_bf16(wqkt + (size_t)(col0 + t * 16) * EMBED + kc, EMBED);
      bv4[t] = load_b_bf16(wvt  + (size_t)(col0 + t * 16) * EMBED + kc, EMBED);
    }
#pragma unroll
    for (int t = 0; t < 4; ++t) accq[t] = wmma_bf16(a, bq[t], accq[t]);
#pragma unroll
    for (int t = 0; t < 4; ++t) accv[t] = wmma_bf16(a, bv4[t], accv[t]);
  }
  unsigned l = lane_id(), nl = l & 15u, h = l >> 4;
#pragma unroll
  for (int t = 0; t < 4; ++t) {
    int c = col0 + t * 16 + (int)nl;
    float bq = bqk[c], bvv = bv[c];
    int hh = c >> 6, hd = c & 63;
    const size_t qb = ((size_t)(bidx * HEADS + hh) * N) * HD + hd;
    const size_t vb0 = ((size_t)(bidx * HEADS + hh) * HD + hd) * (size_t)N;
#pragma unroll
    for (int r = 0; r < 8; ++r) {
      int n = nbase + (int)(h << 3) + r;
      store_bf(qk_out + qb + (size_t)n * HD, accq[t][r] + bq);
      store_bf(vt_out + vb0 + n, accv[t][r] + bvv);
    }
  }
}

// ---------------------------------------------------------------------------
// Kernel 3: flash cross-attention. One wave owns 16 query rows, streams K/V
// in 32-token tiles with online softmax (f32 stats). exp(S) bounces through
// 1KB/wave LDS to convert C-layout -> A-layout for the P·V WMMA.
// Output written as bf16 [B, Nq, 256] (head-merged) for the Wout GEMM.
// ---------------------------------------------------------------------------
__global__ void k_flash(const bf16* __restrict__ qbuf, const bf16* __restrict__ kbuf,
                        const bf16* __restrict__ vtbuf, bf16* __restrict__ mout,
                        int Nq, int Nk, float scale) {
  __shared__ __align__(16) unsigned short plds_u16[4][16 * 32];
  int wave = threadIdx.x >> 5;
  bf16* plds = (bf16*)&plds_u16[wave][0];

  int tile = blockIdx.x * 4 + wave;        // 16-row tile id over B*H*Nq/16
  int tiles_per_bh = Nq >> 4;
  int bh = tile / tiles_per_bh;
  int qt = tile - bh * tiles_per_bh;
  int b = bh >> 2, hh = bh & 3;

  const bf16* Q  = qbuf  + (size_t)bh * Nq * HD + (size_t)qt * 16 * HD;
  const bf16* K  = kbuf  + (size_t)bh * Nk * HD;
  const bf16* VT = vtbuf + (size_t)bh * HD * Nk;

  v16bf qa0 = load_a_bf16(Q, HD);
  v16bf qa1 = load_a_bf16(Q + 32, HD);

  v8f o[4];
#pragma unroll
  for (int t = 0; t < 4; ++t) o[t] = zero8();
  float mx[8], sm[8];
#pragma unroll
  for (int r = 0; r < 8; ++r) { mx[r] = -3.0e38f; sm[r] = 0.f; }

  unsigned l = lane_id(), nl = l & 15u, h = l >> 4;

  for (int j0 = 0; j0 < Nk; j0 += 32) {
    // prefetch next K tile (one 128B row per lane) -> global_prefetch_b8
    if (j0 + 32 < Nk)
      __builtin_prefetch(K + (size_t)(j0 + 32 + (int)l) * HD, 0, 0);

    // preload all 4 K fragments, then burst the QK^T WMMAs
    v16bf kb0 = load_b_bf16(K + (size_t)j0 * HD, HD);
    v16bf kb1 = load_b_bf16(K + (size_t)j0 * HD + 32, HD);
    v16bf kb2 = load_b_bf16(K + (size_t)(j0 + 16) * HD, HD);
    v16bf kb3 = load_b_bf16(K + (size_t)(j0 + 16) * HD + 32, HD);
    v8f s0 = zero8(), s1 = zero8();
    s0 = wmma_bf16(qa0, kb0, s0);
    s0 = wmma_bf16(qa1, kb1, s0);
    s1 = wmma_bf16(qa0, kb2, s1);
    s1 = wmma_bf16(qa1, kb3, s1);

#pragma unroll
    for (int r = 0; r < 8; ++r) {
      float a = s0[r] * scale, c = s1[r] * scale;
      float t = fmaxf(a, c);
#pragma unroll
      for (int mmsk = 8; mmsk >= 1; mmsk >>= 1) t = fmaxf(t, __shfl_xor(t, mmsk, 16));
      float nm = fmaxf(mx[r], t);
      float p0 = __expf(a - nm), p1 = __expf(c - nm);
      float ps = p0 + p1;
#pragma unroll
      for (int mmsk = 8; mmsk >= 1; mmsk >>= 1) ps += __shfl_xor(ps, mmsk, 16);
      float corr = __expf(mx[r] - nm);
      sm[r] = sm[r] * corr + ps;
      mx[r] = nm;
#pragma unroll
      for (int t4 = 0; t4 < 4; ++t4) o[t4][r] *= corr;
      int m = (int)(h << 3) + r;
      store_bf(plds + m * 32 + (int)nl, p0);
      store_bf(plds + m * 32 + (int)nl + 16, p1);
    }
    asm volatile("s_wait_dscnt 0" ::: "memory");   // wave-private LDS bounce

    v16bf pa = load_a_bf16(plds, 32);
    v16bf vb[4];
#pragma unroll
    for (int t4 = 0; t4 < 4; ++t4)
      vb[t4] = load_b_bf16(VT + (size_t)(t4 * 16) * Nk + j0, Nk);
#pragma unroll
    for (int t4 = 0; t4 < 4; ++t4) o[t4] = wmma_bf16(pa, vb[t4], o[t4]);
  }

#pragma unroll
  for (int r = 0; r < 8; ++r) {
    float inv = 1.0f / sm[r];
#pragma unroll
    for (int t4 = 0; t4 < 4; ++t4) o[t4][r] *= inv;
  }
  int nbase = qt * 16;
#pragma unroll
  for (int t4 = 0; t4 < 4; ++t4) {
    int col = hh * HD + t4 * 16 + (int)nl;
#pragma unroll
    for (int r = 0; r < 8; ++r) {
      int m = (int)(h << 3) + r;
      size_t idx = ((size_t)b * Nq + nbase + m) * EMBED + col;
      store_bf(mout + idx, o[t4][r]);
    }
  }
}

// ---------------------------------------------------------------------------
// Kernel 4: 256x256 GEMM (m_raw @ Wout + bout) -> bf16 [rows][256].
// ---------------------------------------------------------------------------
__global__ void k_gemm256(const bf16* __restrict__ A, const bf16* __restrict__ Wt,
                          const float* __restrict__ bias, bf16* __restrict__ out) {
  int wave = threadIdx.x >> 5;
  int row0 = (blockIdx.x * 4 + wave) * 16;
  int col0 = blockIdx.y * 64;
  v8f acc[4];
#pragma unroll
  for (int t = 0; t < 4; ++t) acc[t] = zero8();
  for (int kc = 0; kc < EMBED; kc += 32) {
    v16bf a = load_a_bf16(A + (size_t)row0 * EMBED + kc, EMBED);
    v16bf bfr[4];
#pragma unroll
    for (int t = 0; t < 4; ++t)
      bfr[t] = load_b_bf16(Wt + (size_t)(col0 + t * 16) * EMBED + kc, EMBED);
#pragma unroll
    for (int t = 0; t < 4; ++t) acc[t] = wmma_bf16(a, bfr[t], acc[t]);
  }
  unsigned l = lane_id(), nl = l & 15u, h = l >> 4;
#pragma unroll
  for (int t = 0; t < 4; ++t) {
    int c = col0 + t * 16 + (int)nl;
    float bb = bias[c];
#pragma unroll
    for (int r = 0; r < 8; ++r) {
      int m = (int)(h << 3) + r;
      store_bf(out + (size_t)(row0 + m) * EMBED + c, acc[t][r] + bb);
    }
  }
}

// ---------------------------------------------------------------------------
// Kernel 5: fused FFN: y = [x | m]@W0 + b0; LN; GELU; out = x + y@W3 + b3.
// Block (128 thr, 4 waves) owns 16 rows; y (16x512 f32 = 32KB) lives in LDS
// between the two GEMMs (CDNA5: 320KB LDS/WGP). LN+GELU applied on the fly
// while building the second GEMM's A fragments.
// ---------------------------------------------------------------------------
__global__ void k_ffn(const float* __restrict__ x, const bf16* __restrict__ xb,
                      const bf16* __restrict__ mo,
                      const bf16* __restrict__ w0t, const float* __restrict__ b0,
                      const float* __restrict__ lns, const float* __restrict__ lnb,
                      const bf16* __restrict__ w3t, const float* __restrict__ b3,
                      float* __restrict__ out) {
  __shared__ __align__(16) float y[16 * 512];
  __shared__ float mu[16], rs[16];
  int wave = threadIdx.x >> 5;
  int row0 = blockIdx.x * 16;
  unsigned l = lane_id(), nl = l & 15u, h = l >> 4;

  // Phase A: y = concat(x, m) @ W0 ; wave covers 128 of the 512 columns.
  v8f acc[8];
#pragma unroll
  for (int t = 0; t < 8; ++t) acc[t] = zero8();
  int cbase = wave * 128;
  for (int kc = 0; kc < 512; kc += 32) {
    v16bf a = (kc < 256)
        ? load_a_bf16(xb + (size_t)row0 * EMBED + kc, EMBED)
        : load_a_bf16(mo + (size_t)row0 * EMBED + (kc - 256), EMBED);
    v16bf bfr[8];
#pragma unroll
    for (int t = 0; t < 8; ++t)
      bfr[t] = load_b_bf16(w0t + (size_t)(cbase + t * 16) * 512 + kc, 512);
#pragma unroll
    for (int t = 0; t < 8; ++t) acc[t] = wmma_bf16(a, bfr[t], acc[t]);
  }
#pragma unroll
  for (int t = 0; t < 8; ++t) {
    int c = cbase + t * 16 + (int)nl;
    float bb = b0[c];
#pragma unroll
    for (int r = 0; r < 8; ++r) y[((h << 3) + r) * 512 + c] = acc[t][r] + bb;
  }
  __syncthreads();

  // Phase B: LayerNorm stats; 8 threads per row.
  {
    int r = threadIdx.x >> 3, sub = threadIdx.x & 7;
    float s1 = 0.f, s2 = 0.f;
    for (int c = sub * 64; c < sub * 64 + 64; ++c) {
      float v = y[r * 512 + c];
      s1 += v; s2 += v * v;
    }
#pragma unroll
    for (int mmsk = 4; mmsk >= 1; mmsk >>= 1) {
      s1 += __shfl_xor(s1, mmsk, 8);
      s2 += __shfl_xor(s2, mmsk, 8);
    }
    if (sub == 0) {
      float mean = s1 * (1.0f / 512.0f);
      float var = s2 * (1.0f / 512.0f) - mean * mean;
      mu[r] = mean;
      rs[r] = rsqrtf(var + 1e-5f);
    }
  }
  __syncthreads();

  // Phase C: z = gelu(ln(y)) @ W3 ; wave covers 64 of the 256 out columns.
  v8f z[4];
#pragma unroll
  for (int t = 0; t < 4; ++t) z[t] = zero8();
  int c2 = wave * 64;
  for (int kc = 0; kc < 512; kc += 32) {
    // build A fragment with LN + GELU applied on the fly
    BFrag f;
    {
      unsigned i = nl;
      float mm = mu[i], rr = rs[i];
      const float* row = y + (size_t)i * 512;
#pragma unroll
      for (int v = 0; v < 8; ++v) {
        int k = kc + ((v >> 2) << 4) + (int)(h << 3) + ((v & 3) << 1);
        float a0 = (row[k] - mm) * rr * lns[k] + lnb[k];
        float a1 = (row[k + 1] - mm) * rr * lns[k + 1] + lnb[k + 1];
        f.u[v] = pack2(gelu_exact(a0), gelu_exact(a1));
      }
    }
    v16bf bfr[4];
#pragma unroll
    for (int t = 0; t < 4; ++t)
      bfr[t] = load_b_bf16(w3t + (size_t)(c2 + t * 16) * 512 + kc, 512);
#pragma unroll
    for (int t = 0; t < 4; ++t) z[t] = wmma_bf16(f.v, bfr[t], z[t]);
  }
#pragma unroll
  for (int t = 0; t < 4; ++t) {
    int c = c2 + t * 16 + (int)nl;
    float bb = b3[c];
#pragma unroll
    for (int r = 0; r < 8; ++r) {
      int m = (int)(h << 3) + r;
      size_t idx = (size_t)(row0 + m) * EMBED + c;
      out[idx] = x[idx] + z[t][r] + bb;
    }
  }
}

// ---------------------------------------------------------------------------
// Host launcher
// ---------------------------------------------------------------------------
extern "C" void kernel_launch(void* const* d_in, const int* in_sizes, int n_in,
                              void* d_out, int out_size, void* d_ws, size_t ws_size,
                              hipStream_t stream) {
  const float* x0   = (const float*)d_in[0];
  const float* x1   = (const float*)d_in[1];
  const float* Wqk  = (const float*)d_in[2];
  const float* bqk  = (const float*)d_in[3];
  const float* Wv   = (const float*)d_in[4];
  const float* bv   = (const float*)d_in[5];
  const float* Wout = (const float*)d_in[6];
  const float* bout = (const float*)d_in[7];
  const float* W0   = (const float*)d_in[8];
  const float* b0   = (const float*)d_in[9];
  const float* lns  = (const float*)d_in[10];
  const float* lnb  = (const float*)d_in[11];
  const float* W3   = (const float*)d_in[12];
  const float* b3   = (const float*)d_in[13];

  const int N0 = in_sizes[0] / (BATCH * EMBED);
  const int N1 = in_sizes[1] / (BATCH * EMBED);

  // carve workspace (all 256B-aligned)
  char* ws = (char*)d_ws;
  auto carve = [&](size_t bytes) -> char* {
    char* p = ws;
    ws += (bytes + 255) & ~(size_t)255;
    return p;
  };
  bf16* wqkt  = (bf16*)carve((size_t)EMBED * EMBED * 2);
  bf16* wvt   = (bf16*)carve((size_t)EMBED * EMBED * 2);
  bf16* woutt = (bf16*)carve((size_t)EMBED * EMBED * 2);
  bf16* w0t   = (bf16*)carve((size_t)512 * 512 * 2);
  bf16* w3t   = (bf16*)carve((size_t)EMBED * 512 * 2);
  bf16* xb0   = (bf16*)carve((size_t)BATCH * N0 * EMBED * 2);
  bf16* xb1   = (bf16*)carve((size_t)BATCH * N1 * EMBED * 2);
  bf16* qk0   = (bf16*)carve((size_t)BATCH * EMBED * N0 * 2);
  bf16* qk1   = (bf16*)carve((size_t)BATCH * EMBED * N1 * 2);
  bf16* v0t   = (bf16*)carve((size_t)BATCH * EMBED * N0 * 2);
  bf16* v1t   = (bf16*)carve((size_t)BATCH * EMBED * N1 * 2);
  bf16* m0raw = (bf16*)carve((size_t)BATCH * N0 * EMBED * 2);
  bf16* m1raw = (bf16*)carve((size_t)BATCH * N1 * EMBED * 2);
  bf16* mo0   = (bf16*)carve((size_t)BATCH * N0 * EMBED * 2);
  bf16* mo1   = (bf16*)carve((size_t)BATCH * N1 * EMBED * 2);

  const float scale = 0.125f;  // 1/sqrt(64)

  // 0) x -> bf16 once (removes per-fragment convert VALU from GEMM loops)
  {
    int n40 = BATCH * N0 * EMBED / 4;
    int n41 = BATCH * N1 * EMBED / 4;
    k_cvt<<<(n40 + 255) / 256, 256, 0, stream>>>(x0, xb0, n40);
    k_cvt<<<(n41 + 255) / 256, 256, 0, stream>>>(x1, xb1, n41);
  }

  // 1) weights -> bf16, transposed (K contiguous for B-fragments)
  {
    int n = EMBED * EMBED;
    k_wt<<<(n + 255) / 256, 256, 0, stream>>>(Wqk, wqkt, EMBED, EMBED);
    k_wt<<<(n + 255) / 256, 256, 0, stream>>>(Wv, wvt, EMBED, EMBED);
    k_wt<<<(n + 255) / 256, 256, 0, stream>>>(Wout, woutt, EMBED, EMBED);
    int n0 = 512 * 512;
    k_wt<<<(n0 + 255) / 256, 256, 0, stream>>>(W0, w0t, 512, 512);
    int n3 = 512 * EMBED;
    k_wt<<<(n3 + 255) / 256, 256, 0, stream>>>(W3, w3t, 512, EMBED);
  }

  // 2) projections
  k_project<<<dim3(BATCH * N0 / 64, 4), 128, 0, stream>>>(xb0, wqkt, bqk, wvt, bv,
                                                          qk0, v0t, N0);
  k_project<<<dim3(BATCH * N1 / 64, 4), 128, 0, stream>>>(xb1, wqkt, bqk, wvt, bv,
                                                          qk1, v1t, N1);

  // 3) flash cross-attention, both directions (shared logits, two softmaxes)
  k_flash<<<BATCH * HEADS * N0 / 64, 128, 0, stream>>>(qk0, qk1, v1t, m0raw,
                                                       N0, N1, scale);
  k_flash<<<BATCH * HEADS * N1 / 64, 128, 0, stream>>>(qk1, qk0, v0t, m1raw,
                                                       N1, N0, scale);

  // 4) output projection
  k_gemm256<<<dim3(BATCH * N0 / 64, 4), 128, 0, stream>>>(m0raw, woutt, bout, mo0);
  k_gemm256<<<dim3(BATCH * N1 / 64, 4), 128, 0, stream>>>(m1raw, woutt, bout, mo1);

  // 5) fused FFN (+LN+GELU+residual) -> f32 outputs, concatenated
  float* out0 = (float*)d_out;
  float* out1 = out0 + (size_t)BATCH * N0 * EMBED;
  k_ffn<<<BATCH * N0 / 16, 128, 0, stream>>>(x0, xb0, mo0, w0t, b0, lns, lnb,
                                             w3t, b3, out0);
  k_ffn<<<BATCH * N1 / 16, 128, 0, stream>>>(x1, xb1, mo1, w0t, b0, lns, lnb,
                                             w3t, b3, out1);
}
